// EnhancedContextAwareDualVQ_24902220382570
// MI455X (gfx1250) — matrix operand order repaired and checked by reading, not compiled, not run on hardware
//
#include <hip/hip_runtime.h>
#include <hip/hip_bf16.h>

// Problem constants (fixed by the reference).
#define NROWS 16384     // B*L = 8*2048
#define DDIM  512
#define KCODE 8192
#define KTILE 32        // codes staged per sync window (2 WMMA subtiles)
#define BSTRIDE 520     // LDS row stride in ushorts: 1040B == 16B (mod 256B) -> bank-conflict-free

typedef __attribute__((ext_vector_type(16))) __bf16        v16bf;
typedef __attribute__((ext_vector_type(8)))  float         v8f;
typedef __attribute__((ext_vector_type(8)))  unsigned int  v8u;
typedef __attribute__((vector_size(16)))     int           i32x4;

typedef __attribute__((address_space(3))) char as3_char;

__device__ __forceinline__ unsigned int f2bf_bits(float f) {
  unsigned int u = __float_as_uint(f);
  u += 0x7fffu + ((u >> 16) & 1u);   // round-to-nearest-even into bf16
  return u >> 16;
}
__device__ __forceinline__ unsigned int pk2(float lo, float hi) {
  return f2bf_bits(lo) | (f2bf_bits(hi) << 16);
}

// ---- CDNA5 async global->LDS copy (16B per lane), ASYNCcnt-tracked --------
__device__ __forceinline__ void async_copy16(const void* g, void* l) {
#if __has_builtin(__builtin_amdgcn_global_load_async_to_lds_b128)
  __builtin_amdgcn_global_load_async_to_lds_b128(
      (__attribute__((address_space(1))) i32x4*)g,
      (__attribute__((address_space(3))) i32x4*)l,
      0, 0);
#else
  unsigned lds_off = (unsigned)(size_t)(as3_char*)l;
  asm volatile("global_load_async_to_lds_b128 %0, %1, off"
               :: "v"(lds_off), "v"(g) : "memory");
#endif
}
__device__ __forceinline__ void wait_async0() {
#if __has_builtin(__builtin_amdgcn_s_wait_asynccnt)
  __builtin_amdgcn_s_wait_asynccnt(0);
#else
  asm volatile("s_wait_asynccnt 0" ::: "memory");
#endif
}

// ---- assemble a B fragment (16 bf16 = 32B) from two 16B LDS loads ---------
__device__ __forceinline__ v8u load_frag(const unsigned short* p16) {
  const uint4* p = (const uint4*)p16;
  uint4 lo = p[0];
  uint4 hi = p[1];
  v8u b;
  b[0] = lo.x; b[1] = lo.y; b[2] = lo.z; b[3] = lo.w;
  b[4] = hi.x; b[5] = hi.y; b[6] = hi.z; b[7] = hi.w;
  return b;
}

// ---------------------------------------------------------------------------
// Kernel 1: convert embedding fp32 -> bf16 (row major K x D) and compute
// 0.5*||e_k||^2 per code. One block (256 thr) per code row.
// ---------------------------------------------------------------------------
__global__ __launch_bounds__(256) void vq_prep(const float* __restrict__ E,
                                               unsigned int* __restrict__ ebf_u32,
                                               float* __restrict__ normHalf) {
  const int k = blockIdx.x;
  const int t = threadIdx.x;
  const float* row = E + (size_t)k * DDIM;
  const float a = row[2 * t];
  const float b = row[2 * t + 1];
  ebf_u32[(size_t)k * (DDIM / 2) + t] = pk2(a, b);

  __shared__ float red[256];
  red[t] = a * a + b * b;
  __syncthreads();
  for (int off = 128; off > 0; off >>= 1) {
    if (t < off) red[t] += red[t + off];
    __syncthreads();
  }
  if (t == 0) normHalf[k] = 0.5f * red[0];
}

// ---------------------------------------------------------------------------
// Kernel 2: WMMA distance-argmax with async double-buffered B staging.
// Grid 128 blocks x 256 threads (8 waves). Wave w owns rows [blk*128+w*16,+16).
// A fragments (16x512 bf16) live in registers for the whole K loop.
// Each 32-code B tile (32KB bf16) is staged once per block into LDS via
// global_load_async_to_lds_b128 and shared by all 8 waves; each sync window
// covers two 16x16 WMMA subtiles (32 WMMAs) to amortize barrier cost.
// ---------------------------------------------------------------------------
__global__ __launch_bounds__(256) void vq_argmax(const float* __restrict__ z,
                                                 const unsigned short* __restrict__ ebf,
                                                 const float* __restrict__ normHalf,
                                                 int* __restrict__ idxOut) {
  __shared__ unsigned short btile[2][KTILE * BSTRIDE];   // 2 x 32.5KB (padded)

  const int lane = threadIdx.x & 31;
  const int wave = threadIdx.x >> 5;
  const int half = lane >> 4;    // 0: lanes 0-15, 1: lanes 16-31
  const int l15  = lane & 15;
  const int r0   = blockIdx.x * 128 + wave * 16;

  // staging role: thread t copies 128B = row (t>>3), byte-chunk (t&7)*128
  const int srow    = threadIdx.x >> 3;        // 0..31
  const int schunkB = (threadIdx.x & 7) * 128; // 0..896

  // --- Build A fragments: 16x32 tiles in contiguous-k order (not the ISA table order), same as B.
  // Lane holds 16 contiguous elems: K = h*16..+15.
  v16bf a[16];
  {
    const float* zrow = z + (size_t)(r0 + l15) * DDIM;
#pragma unroll
    for (int s = 0; s < 16; ++s) {
      const float* p0 = zrow + s * 32 + half * 16;  // chunk 0 (8 floats)
      const float* p1 = p0 + 8;                     // chunk 1 (8 floats)
      float4 f0 = *(const float4*)(p0);
      float4 f1 = *(const float4*)(p0 + 4);
      float4 f2 = *(const float4*)(p1);
      float4 f3 = *(const float4*)(p1 + 4);
      v8u au;
      au[0] = pk2(f0.x, f0.y); au[1] = pk2(f0.z, f0.w);
      au[2] = pk2(f1.x, f1.y); au[3] = pk2(f1.z, f1.w);
      au[4] = pk2(f2.x, f2.y); au[5] = pk2(f2.z, f2.w);
      au[6] = pk2(f3.x, f3.y); au[7] = pk2(f3.z, f3.w);
      a[s] = __builtin_bit_cast(v16bf, au);
    }
  }

  float best[8];
  int   bidx[8];
#pragma unroll
  for (int i = 0; i < 8; ++i) { best[i] = -3.402823466e38f; bidx[i] = 0; }

  // --- stage first tile (kt = 0): 32 rows x 1KB
  {
    const char* g = (const char*)(ebf + ((size_t)srow << 9)) + schunkB;
    char* l = (char*)(&btile[0][(size_t)srow * BSTRIDE]) + schunkB;
#pragma unroll
    for (int j = 0; j < 8; ++j) async_copy16(g + j * 16, l + j * 16);
  }
  wait_async0();
  __syncthreads();

  for (int kt = 0; kt < KCODE; kt += KTILE) {
    const int buf = (kt >> 5) & 1;

    // prefetch next tile into the other buffer (async, overlapped with WMMAs)
    if (kt + KTILE < KCODE) {
      const char* g = (const char*)(ebf + ((size_t)(kt + KTILE + srow) << 9)) + schunkB;
      char* l = (char*)(&btile[buf ^ 1][(size_t)srow * BSTRIDE]) + schunkB;
#pragma unroll
      for (int j = 0; j < 8; ++j) async_copy16(g + j * 16, l + j * 16);
    }

#pragma unroll
    for (int sub = 0; sub < 2; ++sub) {
      const int col = kt + sub * 16 + l15;  // code handled by this lane (N dim)
      const float cnorm = normHalf[col];    // 0.5*||e||^2; waited on only at the end

      // B fragments from LDS: lane reads 32B of code row (sub*16+l15).
      const unsigned short* brow =
          &btile[buf][(size_t)(sub * 16 + l15) * BSTRIDE + half * 16];

      v8f acc = {};                         // zero C folds into WMMA inline src2
      v8u bu[2];
      bu[0] = load_frag(brow);
#pragma unroll
      for (int s = 0; s < 16; ++s) {
        if (s < 15) bu[(s + 1) & 1] = load_frag(brow + (s + 1) * 32);
        acc = __builtin_amdgcn_wmma_f32_16x16x32_bf16(
            false, a[s], false, __builtin_bit_cast(v16bf, bu[s & 1]),
            (short)0, acc, false, false);
      }

      // score = dot - 0.5||e||^2; strict '>' keeps lowest index on ties.
#pragma unroll
      for (int i = 0; i < 8; ++i) {
        const float sc = acc[i] - cnorm;
        if (sc > best[i]) { best[i] = sc; bidx[i] = col; }
      }
    }

    wait_async0();      // my async stage for the next tile is complete
    __syncthreads();    // everyone staged + everyone done reading this tile
  }

  // Butterfly reduce across the 16 lanes holding the same row.
#pragma unroll
  for (int m = 8; m >= 1; m >>= 1) {
#pragma unroll
    for (int i = 0; i < 8; ++i) {
      float ov = __shfl_xor(best[i], m, 32);
      int   oi = __shfl_xor(bidx[i], m, 32);
      if (ov > best[i] || (ov == best[i] && oi < bidx[i])) {
        best[i] = ov; bidx[i] = oi;
      }
    }
  }
  // Row of acc[i] is M = i + 8*half. Lane 0 / lane 16 publish winners.
  if (l15 == 0) {
#pragma unroll
    for (int i = 0; i < 8; ++i) idxOut[r0 + half * 8 + i] = bidx[i];
  }
}

// ---------------------------------------------------------------------------
// Kernel 3: gather z_quant = embedding[idx] (fp32) and per-row loss partials.
// One block (256 thr) per row; fixed-order tree reduction (deterministic).
// ---------------------------------------------------------------------------
__global__ __launch_bounds__(256) void vq_gather(const float* __restrict__ z,
                                                 const float* __restrict__ E,
                                                 const int* __restrict__ idx,
                                                 float* __restrict__ zq,
                                                 float* __restrict__ rowloss) {
  const int row = blockIdx.x;
  const int t = threadIdx.x;
  const int c = idx[row];
  const float2* e  = (const float2*)(E + (size_t)c * DDIM);
  const float2* zr = (const float2*)(z + (size_t)row * DDIM);
  float2* o = (float2*)(zq + (size_t)row * DDIM);

  float2 ev = e[t];
  float2 zv = zr[t];
  o[t] = ev;
  const float dx = ev.x - zv.x;
  const float dy = ev.y - zv.y;

  __shared__ float red[256];
  red[t] = dx * dx + dy * dy;
  __syncthreads();
  for (int off = 128; off > 0; off >>= 1) {
    if (t < off) red[t] += red[t + off];
    __syncthreads();
  }
  if (t == 0) rowloss[row] = red[0];
}

// ---------------------------------------------------------------------------
// Kernel 4: final deterministic loss reduction -> vq_loss scalar.
// ---------------------------------------------------------------------------
__global__ __launch_bounds__(256) void vq_loss_final(const float* __restrict__ rowloss,
                                                     float* __restrict__ lossOut) {
  const int t = threadIdx.x;
  float s = 0.0f;
  for (int r = t; r < NROWS; r += 256) s += rowloss[r];
  __shared__ float red[256];
  red[t] = s;
  __syncthreads();
  for (int off = 128; off > 0; off >>= 1) {
    if (t < off) red[t] += red[t + off];
    __syncthreads();
  }
  if (t == 0) {
    lossOut[0] = 2.0f * red[0] / ((float)NROWS * (float)DDIM);
  }
}

// ---------------------------------------------------------------------------
extern "C" void kernel_launch(void* const* d_in, const int* in_sizes, int n_in,
                              void* d_out, int out_size, void* d_ws, size_t ws_size,
                              hipStream_t stream) {
  const float* z = (const float*)d_in[0];   // (8,2048,512) fp32
  const float* E = (const float*)d_in[1];   // (8192,512)  fp32
  float* out = (float*)d_out;               // z_quant (8.4M) ++ vq_loss (1)

  char* ws = (char*)d_ws;
  // ws layout (bytes):
  //   [0,            8388608)  : E in bf16 (8192*512*2)
  //   [8388608,      8421376)  : 0.5*||e_k||^2  (8192 f32)
  //   [8421376,      8486912)  : argmin indices (16384 i32)
  //   [8486912,      8552448)  : per-row loss partials (16384 f32)
  unsigned short* ebf      = (unsigned short*)(ws);
  unsigned int*   ebf_u32  = (unsigned int*)(ws);
  float*          normHalf = (float*)(ws + 8388608);
  int*            idx      = (int*)(ws + 8421376);
  float*          rowloss  = (float*)(ws + 8486912);

  vq_prep<<<KCODE, 256, 0, stream>>>(E, ebf_u32, normHalf);
  vq_argmax<<<NROWS / 128, 256, 0, stream>>>(z, ebf, normHalf, idx);
  vq_gather<<<NROWS, 256, 0, stream>>>(z, E, idx, out, rowloss);
  vq_loss_final<<<1, 256, 0, stream>>>(rowloss, out + (size_t)NROWS * DDIM);
}
